// DecomposedSTFT_28973849379568
// MI455X (gfx1250) — matrix operand order repaired and checked by reading, not compiled, not run on hardware
//
#include <hip/hip_runtime.h>

// STFT power spectrum as a WMMA GEMM:
//   P[1026,30001] = Basis[1026,1024] x X[1024,30001],  X[k,t] = reflect_pad(audio)[t*512+k]
//   out[f,t] = P[f,t]^2 + P[f+513,t]^2   (f in 0..512)
// f32 accuracy on the f16 WMMA path via hi/lo split (hi*hi + hi*lo + lo*hi).
// The split basis is pre-swizzled into the exact WMMA A-fragment layout in d_ws
// so the hot loop has ZERO conversion VALU: just b128 loads + v_wmma.

#define FILTER_LENGTH 1024
#define HOP           512
#define CUTOFF        513                        // FILTER_LENGTH/2 + 1
#define PAD           512
#define NFT           ((CUTOFF + 15) / 16)       // 33 frequency tiles
#define NKC           (FILTER_LENGTH / 32)       // 32 K-chunks
#define FRAMES_PER_BLK 32                        // two 16-wide N tiles per wave
#define WIN_ELEMS     ((FRAMES_PER_BLK - 1) * HOP + FILTER_LENGTH)  // 16896
#define FRAG_HALFS    ((size_t)NFT * NKC * 32 * 16)                 // halfs per packed array

typedef __attribute__((ext_vector_type(16))) _Float16 v16h;
typedef __attribute__((ext_vector_type(8)))  float    v8f;

// ---------------------------------------------------------------------------
// Prep: split basis f32 -> (hi,lo) f16 and store in WMMA A-fragment order.
// Fragment k order used here (16-bit A 16x32, contiguous, B matches): lane half 0 -> K {0..15},
// half 1 -> K {16..31}; element e -> ka+e.
// Packed address: ((ft*NKC + kc)*32 + lane)*16 + e
// ---------------------------------------------------------------------------
__global__ __launch_bounds__(32) void pack_basis(
    const float* __restrict__ basis,
    _Float16* __restrict__ Rhi, _Float16* __restrict__ Rlo,
    _Float16* __restrict__ Ihi, _Float16* __restrict__ Ilo)
{
  const int lane = threadIdx.x;          // 0..31 (one wave)
  const int ft   = blockIdx.x;           // 0..NFT-1
  const int kc   = blockIdx.y;           // 0..NKC-1
  const int m    = lane & 15;
  const int hf   = lane >> 4;

  int rowR = ft * 16 + m; if (rowR > CUTOFF - 1) rowR = CUTOFF - 1;
  const float* __restrict__ pR = basis + (long)rowR * FILTER_LENGTH;
  const float* __restrict__ pI = pR + (long)CUTOFF * FILTER_LENGTH;

  const int  ka  = kc * 32 + hf * 16;
  const long off = (((long)(ft * NKC + kc)) * 32 + lane) * 16;

#pragma unroll
  for (int e = 0; e < 16; ++e) {
    const int k = ka + e;
    float v = pR[k];
    _Float16 h = (_Float16)v;
    Rhi[off + e] = h;
    Rlo[off + e] = (_Float16)(v - (float)h);
    v = pI[k];
    h = (_Float16)v;
    Ihi[off + e] = h;
    Ilo[off + e] = (_Float16)(v - (float)h);
  }
}

// ---------------------------------------------------------------------------
// Main kernel: 256 threads = 8 waves. Block owns 32 frames; each wave owns one
// 16-row frequency tile and two 16-column N tiles.
// ---------------------------------------------------------------------------
__global__ __launch_bounds__(256) void stft_pow_wmma(
    const float* __restrict__ audio,
    const _Float16* __restrict__ Rhi, const _Float16* __restrict__ Rlo,
    const _Float16* __restrict__ Ihi, const _Float16* __restrict__ Ilo,
    float* __restrict__ out,
    int nAudio, int nFrames)
{
  // Audio window staged pre-split into f16 hi/lo (66 KB LDS).
  __shared__ _Float16 sHi[WIN_ELEMS];
  __shared__ _Float16 sLo[WIN_ELEMS];

  const int  t0    = blockIdx.x * FRAMES_PER_BLK;
  const long basep = (long)t0 * HOP - PAD;

  // 16896 = 66 * 256 -> perfectly uniform staging loop.
  for (int s = threadIdx.x; s < WIN_ELEMS; s += 256) {
    long j = basep + s;
    j = (j < 0) ? -j : j;                          // reflect left
    if (j >= nAudio) j = 2L * (nAudio - 1) - j;    // reflect right
    float v = audio[j];
    _Float16 h = (_Float16)v;
    sHi[s] = h;
    sLo[s] = (_Float16)(v - (float)h);
  }
  __syncthreads();

  const int lane = threadIdx.x & 31;
  const int wave = threadIdx.x >> 5;
  const int ft   = blockIdx.y * 8 + wave;
  if (ft >= NFT) return;                           // after barrier: safe

  const int fr0 = ft * 16;
  const int m   = lane & 15;
  const int hf  = lane >> 4;

  // Packed A-fragment pointers for this (ft, lane); chunk stride = 32*16 halfs.
  const long a0 = (((long)ft * NKC) * 32 + lane) * 16;
  const _Float16* __restrict__ pRh = Rhi + a0;
  const _Float16* __restrict__ pRl = Rlo + a0;
  const _Float16* __restrict__ pIh = Ihi + a0;
  const _Float16* __restrict__ pIl = Ilo + a0;

  // B fragment bases in LDS for the two N tiles (32-byte aligned).
  const int bb0 = m * HOP + hf * 16;
  const int bb1 = (m + 16) * HOP + hf * 16;

  v8f accR0 = {}, accI0 = {}, accR1 = {}, accI1 = {};

  for (int kc = 0; kc < NKC; ++kc) {
    const int kb = kc * 32;
    const long ao = (long)kc * (32 * 16);

    // A fragments: pure 32B vector loads, no conversion VALU.
    v16h aRh = *(const v16h*)(pRh + ao);
    v16h aRl = *(const v16h*)(pRl + ao);
    v16h aIh = *(const v16h*)(pIh + ao);
    v16h aIl = *(const v16h*)(pIl + ao);

    // B fragments (shared real/imag) straight from LDS.
    v16h b0h = *(const v16h*)&sHi[bb0 + kb];
    v16h b0l = *(const v16h*)&sLo[bb0 + kb];
    v16h b1h = *(const v16h*)&sHi[bb1 + kb];
    v16h b1l = *(const v16h*)&sLo[bb1 + kb];

    // hi*hi + hi*lo + lo*hi per (real,imag) x (Ntile0, Ntile1): 12 WMMAs.
    accR0 = __builtin_amdgcn_wmma_f32_16x16x32_f16(false, aRh, false, b0h, (short)0, accR0, false, false);
    accR0 = __builtin_amdgcn_wmma_f32_16x16x32_f16(false, aRh, false, b0l, (short)0, accR0, false, false);
    accR0 = __builtin_amdgcn_wmma_f32_16x16x32_f16(false, aRl, false, b0h, (short)0, accR0, false, false);
    accI0 = __builtin_amdgcn_wmma_f32_16x16x32_f16(false, aIh, false, b0h, (short)0, accI0, false, false);
    accI0 = __builtin_amdgcn_wmma_f32_16x16x32_f16(false, aIh, false, b0l, (short)0, accI0, false, false);
    accI0 = __builtin_amdgcn_wmma_f32_16x16x32_f16(false, aIl, false, b0h, (short)0, accI0, false, false);
    accR1 = __builtin_amdgcn_wmma_f32_16x16x32_f16(false, aRh, false, b1h, (short)0, accR1, false, false);
    accR1 = __builtin_amdgcn_wmma_f32_16x16x32_f16(false, aRh, false, b1l, (short)0, accR1, false, false);
    accR1 = __builtin_amdgcn_wmma_f32_16x16x32_f16(false, aRl, false, b1h, (short)0, accR1, false, false);
    accI1 = __builtin_amdgcn_wmma_f32_16x16x32_f16(false, aIh, false, b1h, (short)0, accI1, false, false);
    accI1 = __builtin_amdgcn_wmma_f32_16x16x32_f16(false, aIh, false, b1l, (short)0, accI1, false, false);
    accI1 = __builtin_amdgcn_wmma_f32_16x16x32_f16(false, aIl, false, b1h, (short)0, accI1, false, false);
  }

  // D layout (f32 16x16): lane L, VGPR r -> M = (L>>4)*8 + r, N = L&15.
  const int col0 = t0 + m;
  const int col1 = t0 + 16 + m;
#pragma unroll
  for (int r = 0; r < 8; ++r) {
    const int row = fr0 + hf * 8 + r;
    if (row < CUTOFF) {
      const long rb = (long)row * nFrames;
      if (col0 < nFrames) {
        float re = accR0[r], im = accI0[r];
        out[rb + col0] = re * re + im * im;
      }
      if (col1 < nFrames) {
        float re = accR1[r], im = accI1[r];
        out[rb + col1] = re * re + im * im;
      }
    }
  }
}

extern "C" void kernel_launch(void* const* d_in, const int* in_sizes, int n_in,
                              void* d_out, int out_size, void* d_ws, size_t ws_size,
                              hipStream_t stream) {
  const float* audio = (const float*)d_in[0];
  const float* basis = (const float*)d_in[1];
  float* out = (float*)d_out;

  const int nAudio  = in_sizes[0];          // 15,360,000
  const int nFrames = nAudio / HOP + 1;     // 30,001

  // Packed split-basis fragments in workspace (4 arrays, ~4.3 MB total).
  _Float16* Rhi = (_Float16*)d_ws;
  _Float16* Rlo = Rhi + FRAG_HALFS;
  _Float16* Ihi = Rlo + FRAG_HALFS;
  _Float16* Ilo = Ihi + FRAG_HALFS;

  pack_basis<<<dim3(NFT, NKC), dim3(32, 1, 1), 0, stream>>>(basis, Rhi, Rlo, Ihi, Ilo);

  dim3 grid((nFrames + FRAMES_PER_BLK - 1) / FRAMES_PER_BLK, (NFT + 7) / 8);  // 938 x 5
  stft_pow_wmma<<<grid, dim3(256, 1, 1), 0, stream>>>(
      audio, Rhi, Rlo, Ihi, Ilo, out, nAudio, nFrames);
}
